// GlobalState_72636486910338
// MI455X (gfx1250) — matrix operand order repaired and checked by reading, not compiled, not run on hardware
//
#include <hip/hip_runtime.h>
#include <hip/hip_bf16.h>
#include <math.h>

typedef __attribute__((ext_vector_type(16))) _Float16 v16h;
typedef __attribute__((ext_vector_type(8)))  _Float16 v8h;
typedef __attribute__((ext_vector_type(4)))  _Float16 v4h;
typedef __attribute__((ext_vector_type(8)))  float    v8f;
typedef __attribute__((ext_vector_type(4)))  float    v4f;

static constexpr int HD = 128;    // hidden/feature dim
static constexpr int TD = 1024;   // time steps
static constexpr int BD = 512;    // batch
static constexpr int G3 = 384;    // 3*H gate columns

__device__ __forceinline__ float fast_sigmoid(float x) {
  return 1.0f / (1.0f + __expf(-x));
}
__device__ __forceinline__ float fast_tanh(float x) {
  return 2.0f / (1.0f + __expf(-2.0f * x)) - 1.0f;
}

// Persistent GRU scan: one workgroup owns 16 batch rows for all T steps.
// 24 waves; wave w owns gate-column tile [16w, 16w+16).
__global__ __launch_bounds__(768) void gru_scan_kernel(
    const float* __restrict__ x,      // (B, T, 128)
    const float* __restrict__ Wk,     // (128, 384)  input kernel
    const float* __restrict__ Wr,     // (128, 384)  recurrent kernel
    const float* __restrict__ bias,   // (2, 384)
    float* __restrict__ h_out)        // (B, 128)
{
  __shared__ _Float16 sh_x [2][16 * HD];  // double-buffered x_t tile, f16 (A operand)
  __shared__ _Float16 sh_h [16 * HD];     // h tile, f16 (A operand)
  __shared__ float    sh_hf[16 * HD];     // h tile, f32 (exact elementwise state)
  __shared__ float    S_x  [16 * G3];     // staged input-side preacts  (16 x 384)
  __shared__ float    S_h  [16 * G3];     // staged recur-side preacts  (16 x 384)

  const int tid  = threadIdx.x;
  const int wave = tid >> 5;           // 0..23
  const int lane = tid & 31;
  const int n    = lane & 15;          // B/C/D column within tile
  const int half = lane >> 4;          // lane-half per ISA layouts
  const int col0 = wave * 16;          // gate-column tile base (0..368)
  const int b0   = blockIdx.x * 16;    // batch row block

  // ---- Preload weight B-fragments into VGPRs (held for all 1024 steps) ----
  // B (32x16) in this code's k order: lanes 0-15 hold K=0..15, lanes 16-31 hold K=16..31 (A matches).
  v16h bk[4], bu[4];
#pragma unroll
  for (int c = 0; c < 4; ++c) {
    v16h tk, tu;
#pragma unroll
    for (int e = 0; e < 16; ++e) {
      int k = c * 32 + half * 16 + e;
      tk[e] = (_Float16)Wk[k * G3 + col0 + n];
      tu[e] = (_Float16)Wr[k * G3 + col0 + n];
    }
    bk[c] = tk; bu[c] = tu;
  }

  // ---- Hoist bias into registers (gate stage: threads 0..511, 4 cols each) ----
  const int e0 = tid * 4;              // element base for gate/prefetch stages
  const int gc = e0 & 127;             // column base (multiple of 4)
  v4f bz = {}, br = {}, bxh = {}, bhu = {};
  if (tid < 512) {
    v4f b0z = *(const v4f*)(bias + gc);
    v4f b1z = *(const v4f*)(bias + G3 + gc);
    v4f b0r = *(const v4f*)(bias + 128 + gc);
    v4f b1r = *(const v4f*)(bias + G3 + 128 + gc);
    bxh     = *(const v4f*)(bias + 256 + gc);
    bhu     = *(const v4f*)(bias + G3 + 256 + gc);
    bz = b0z + b1z;
    br = b0r + b1r;
  }

  // per-thread global source pointer for the x stream (rows of this block's tile)
  const float* xsrc = x + (size_t)(b0 + (e0 >> 7)) * (TD * HD) + (e0 & 127);

  // ---- h0 = 0, and stage x_0 into buffer 0 (vectorized float4 -> 4xf16) ----
  for (int idx = tid; idx < 16 * HD; idx += (int)blockDim.x) {
    sh_hf[idx] = 0.0f;
    sh_h [idx] = (_Float16)0.0f;
  }
  if (tid < 512) {
    v4f v = *(const v4f*)xsrc;
    v4h p; p[0] = (_Float16)v.x; p[1] = (_Float16)v.y;
           p[2] = (_Float16)v.z; p[3] = (_Float16)v.w;
    *(v4h*)(&sh_x[0][e0]) = p;
  }
  __syncthreads();

  const int m = n;  // A-matrix row owned by this lane (both halves share M)

  for (int t = 0; t < TD; ++t) {
    const int p = t & 1;

    // ---- issue the global load for x_{t+1} NOW (wait is deferred past the WMMAs) ----
    // clamped index avoids loop peeling; last iteration harmlessly re-reads x_{T-1}
    const int tn = (t + 1 < TD) ? (t + 1) : (TD - 1);
    v4f xv = {};
    if (tid < 512) {
      xv = *(const v4f*)(xsrc + (size_t)tn * HD);
    }

    // ---- two fused GEMM tiles: x_t @ Wk[:,tile]  and  h @ Wr[:,tile] ----
    v8f xacc = {};
    v8f hacc = {};
#pragma unroll
    for (int c = 0; c < 4; ++c) {
      // 16-bit A (16x32): half 0 -> K {c*32+0..15}; half 1 -> {+16..31} (matches B)
      const int kb = c * 32 + half * 16;
      v8h xlo = *(const v8h*)(&sh_x[p][m * HD + kb]);
      v8h xhi = *(const v8h*)(&sh_x[p][m * HD + kb + 8]);
      v8h hlo = *(const v8h*)(sh_h + m * HD + kb);
      v8h hhi = *(const v8h*)(sh_h + m * HD + kb + 8);
      v16h ax, ah;
#pragma unroll
      for (int e = 0; e < 8; ++e) {
        ax[e] = xlo[e]; ax[8 + e] = xhi[e];
        ah[e] = hlo[e]; ah[8 + e] = hhi[e];
      }
      xacc = __builtin_amdgcn_wmma_f32_16x16x32_f16(false, ax, false, bk[c],
                                                    (short)0, xacc, false, false);
      hacc = __builtin_amdgcn_wmma_f32_16x16x32_f16(false, ah, false, bu[c],
                                                    (short)0, hacc, false, false);
    }

    // ---- stage pre-activations, branch-free (C/D layout: VGPR r -> M = half*8 + r) ----
#pragma unroll
    for (int r = 0; r < 8; ++r) {
      const int row = half * 8 + r;
      S_x[row * G3 + col0 + n] = xacc[r];
      S_h[row * G3 + col0 + n] = hacc[r];
    }

    // ---- now land the prefetched x_{t+1} into the back buffer (load wait sits here) ----
    if (tid < 512) {
      v4h pk; pk[0] = (_Float16)xv.x; pk[1] = (_Float16)xv.y;
              pk[2] = (_Float16)xv.z; pk[3] = (_Float16)xv.w;
      *(v4h*)(&sh_x[1 - p][e0]) = pk;
    }
    __syncthreads();

    // ---- elementwise gate update: 512 threads x 4 consecutive columns, all vector ops ----
    if (tid < 512) {
      const int mm   = e0 >> 7;
      const int base = mm * G3 + gc;
      v4f sxz = *(const v4f*)(S_x + base);
      v4f shz = *(const v4f*)(S_h + base);
      v4f sxr = *(const v4f*)(S_x + base + 128);
      v4f shr = *(const v4f*)(S_h + base + 128);
      v4f sxh = *(const v4f*)(S_x + base + 256);
      v4f shu = *(const v4f*)(S_h + base + 256);
      v4f hold = *(const v4f*)(sh_hf + e0);
      v4f hn;
      v4h h16;
#pragma unroll
      for (int j = 0; j < 4; ++j) {
        float z  = fast_sigmoid(sxz[j] + shz[j] + bz[j]);
        float r  = fast_sigmoid(sxr[j] + shr[j] + br[j]);
        float hh = fast_tanh(sxh[j] + bxh[j] + r * (shu[j] + bhu[j]));
        float v  = z * hold[j] + (1.0f - z) * hh;
        hn[j]  = v;
        h16[j] = (_Float16)v;
      }
      *(v4f*)(sh_hf + e0) = hn;
      *(v4h*)(sh_h + e0)  = h16;
    }
    __syncthreads();
  }

  // ---- write final hidden state ----
  for (int idx = tid; idx < 16 * HD; idx += (int)blockDim.x) {
    const int row = idx >> 7, f = idx & 127;
    h_out[(size_t)(b0 + row) * HD + f] = sh_hf[idx];
  }
}

// Tiny MLP head: one block (128 threads) per batch row.
__global__ __launch_bounds__(128) void gru_head_kernel(
    const float* __restrict__ hbuf,  // (B, 128)
    const float* __restrict__ w2, const float* __restrict__ b2,   // (128,128), (128)
    const float* __restrict__ w3, const float* __restrict__ b3,   // (128,64),  (64)
    const float* __restrict__ wo, const float* __restrict__ bo,   // (64,1),    (1)
    float* __restrict__ out)         // (B, 1)
{
  __shared__ float hr[128];
  __shared__ float v1[128];
  __shared__ float v2[64];
  const int tid = threadIdx.x;
  const int b   = blockIdx.x;

  hr[tid] = hbuf[(size_t)b * 128 + tid];
  __syncthreads();

  float a1 = b2[tid];
#pragma unroll 8
  for (int k = 0; k < 128; ++k) a1 += hr[k] * w2[k * 128 + tid];
  v1[tid] = fmaxf(a1, 0.0f);
  __syncthreads();

  if (tid < 64) {
    float a2 = b3[tid];
#pragma unroll 8
    for (int k = 0; k < 128; ++k) a2 += v1[k] * w3[k * 64 + tid];
    v2[tid] = fmaxf(a2, 0.0f);
  }
  __syncthreads();

  if (tid == 0) {
    float a3 = bo[0];
#pragma unroll 8
    for (int k = 0; k < 64; ++k) a3 += v2[k] * wo[k];
    out[b] = 1.0f / (1.0f + expf(-a3));
  }
}

extern "C" void kernel_launch(void* const* d_in, const int* in_sizes, int n_in,
                              void* d_out, int out_size, void* d_ws, size_t ws_size,
                              hipStream_t stream) {
  (void)in_sizes; (void)n_in; (void)out_size; (void)ws_size;
  const float* x    = (const float*)d_in[0];
  const float* Wk   = (const float*)d_in[1];
  const float* Wr   = (const float*)d_in[2];
  const float* bias = (const float*)d_in[3];
  const float* w2   = (const float*)d_in[4];
  const float* b2   = (const float*)d_in[5];
  const float* w3   = (const float*)d_in[6];
  const float* b3   = (const float*)d_in[7];
  const float* wo   = (const float*)d_in[8];
  const float* bo   = (const float*)d_in[9];

  float* hbuf = (float*)d_ws;   // 512*128 floats = 256 KB scratch for final h

  gru_scan_kernel<<<BD / 16, 768, 0, stream>>>(x, Wk, Wr, bias, hbuf);
  gru_head_kernel<<<BD, 128, 0, stream>>>(hbuf, w2, b2, w3, b3, wo, bo, (float*)d_out);
}
